// CL_model_15960098472226
// MI455X (gfx1250) — matrix operand order repaired and checked by reading, not compiled, not run on hardware
//
#include <hip/hip_runtime.h>
#include <cstddef>

// ---------------- problem constants ----------------
#define Bz   4096
#define Sz   50
#define Dz   128
#define Hz   4
#define DHz  32
#define Iz   256
#define Lz   4
#define MTOK (Bz*Sz)          // 204800 padded token rows
#define NEGV (-100000.0f)

typedef _Float16 h16;
typedef h16  v8h  __attribute__((ext_vector_type(8)));
typedef h16  v16h __attribute__((ext_vector_type(16)));
typedef float v8f __attribute__((ext_vector_type(8)));
typedef unsigned int v4u __attribute__((ext_vector_type(4)));
typedef int v8i __attribute__((ext_vector_type(8)));
typedef int v4i __attribute__((ext_vector_type(4)));

// build a 16-half WMMA fragment from two contiguous 8-half (16B) chunks
static __device__ inline v16h mk16(const h16* p0, const h16* p1) {
  v8h lo = *(const v8h*)p0, hi = *(const v8h*)p1;
  v16h r;
#pragma unroll
  for (int i = 0; i < 8; ++i) { r[i] = lo[i]; r[i + 8] = hi[i]; }
  return r;
}

static __device__ inline v8f wmma_f16(v16h a, v16h b, v8f c) {
  // 8 args: (neg_a, A, neg_b, B, c_mod, C, reuse_a, reuse_b)
  return __builtin_amdgcn_wmma_f32_16x16x32_f16(false, a, false, b, (short)0, c, false, false);
}

// ---- Tensor Data Mover: DMA a 64-row x 128-half tile (row stride = strideElems halfs)
// from global into LDS, inserting 16B of LDS padding per row (64-dword interval -> code 5,
// 4-dword pad amount -> code 3), i.e. LDS row pitch = 136 halfs.
static __device__ inline void tdm_load_tile(unsigned lds_addr, const void* gptr, int strideElems) {
  unsigned long long ga = (unsigned long long)(size_t)gptr;
  v4u g0;
  g0[0] = 1u;                                   // count=1 (valid user D#), no gather
  g0[1] = lds_addr;                             // LDS byte address
  g0[2] = (unsigned)ga;                         // global_addr[31:0]
  g0[3] = (unsigned)(ga >> 32) | (2u << 30);    // global_addr[56:32] | type=2 ("image")
  v8i g1;
  g1[0] = (int)((1u << 16)                      // data_size = 2 bytes
              | (1u << 20)                      // pad_enable
              | (5u << 22)                      // pad_interval: 64 dwords (=128 halfs)
              | (3u << 25));                    // pad_amount: 4 dwords (=8 halfs)
  g1[1] = (int)(128u << 16);                    // tensor_dim0 = 128 (elements)
  g1[2] = (int)(64u << 16);                     // tensor_dim0 hi | tensor_dim1 = 64
  g1[3] = (int)(128u << 16);                    // tensor_dim1 hi | tile_dim0 = 128
  g1[4] = 64;                                   // tile_dim1 = 64, tile_dim2 = 0
  g1[5] = strideElems;                          // tensor_dim0_stride[31:0]
  g1[6] = 0;                                    // stride hi | dim1_stride lo (unused, tile_dim2=0)
  g1[7] = 0;
  v4i z4 = {0, 0, 0, 0};
#if defined(__clang_major__) && (__clang_major__ >= 23)
  v8i z8 = {0, 0, 0, 0, 0, 0, 0, 0};
  __builtin_amdgcn_tensor_load_to_lds(g0, g1, z4, z4, z8, 0);
#else
  __builtin_amdgcn_tensor_load_to_lds(g0, g1, z4, z4, 0);
#endif
}

// ---------------- scatter/pad helpers ----------------
__global__ void k_count(const int* __restrict__ batch, int* __restrict__ counts, int n) {
  int i = blockIdx.x * blockDim.x + threadIdx.x;
  if (i < n) atomicAdd(&counts[batch[i]], 1);
}

__global__ void k_scan(const int* __restrict__ counts, int* __restrict__ starts, int* __restrict__ cc) {
  if (threadIdx.x == 0 && blockIdx.x == 0) {
    int acc = 0;
    for (int b = 0; b < Bz; ++b) {
      starts[b] = acc; acc += counts[b];
      cc[b] = counts[b] < Sz ? counts[b] : Sz;
    }
  }
}

__global__ void k_nodemap(const int* __restrict__ batch, const int* __restrict__ starts,
                          int* __restrict__ nodemap, int n) {
  int i = blockIdx.x * blockDim.x + threadIdx.x;
  if (i < n) {
    int b = batch[i];
    int pos = i - starts[b];
    if (pos < Sz) nodemap[b * Sz + pos] = i;
  }
}

__global__ __launch_bounds__(128) void k_embed2d(
    const int* __restrict__ nodemap, const float* __restrict__ emb,
    const float* __restrict__ din, const float* __restrict__ dout,
    const int* __restrict__ ideg, const int* __restrict__ odeg,
    float* __restrict__ hf, h16* __restrict__ hh) {
  int row = blockIdx.x, d = threadIdx.x;
  int i = nodemap[row];
  int a = (i >= 0) ? ideg[i] : 0;     // padded rows hit index 0, like the reference scatter
  int o = (i >= 0) ? odeg[i] : 0;
  float v = din[a * Dz + d] + dout[o * Dz + d];
  if (i >= 0) v += emb[(size_t)i * Dz + d];
  hf[(size_t)row * Dz + d] = v;
  hh[(size_t)row * Dz + d] = (h16)v;
}

__global__ __launch_bounds__(128) void k_embed3d(
    const int* __restrict__ nodemap, const float* __restrict__ emb,
    const float* __restrict__ d3, const float* __restrict__ dw,
    float* __restrict__ hf, h16* __restrict__ hh) {
  int row = blockIdx.x, d = threadIdx.x;
  int i = nodemap[row];
  float v = (i >= 0) ? (emb[(size_t)i * Dz + d] + d3[i] * dw[d]) : 0.f;
  hf[(size_t)row * Dz + d] = v;
  hh[(size_t)row * Dz + d] = (h16)v;
}

// convert f32 [K][N] weight -> f16 transposed [N][K]
__global__ void k_cvt(const float* __restrict__ w, h16* __restrict__ wt, int K, int N) {
  int i = blockIdx.x * blockDim.x + threadIdx.x;
  if (i < K * N) { int k = i / N, n = i % N; wt[(size_t)n * K + k] = (h16)w[i]; }
}

// ---------------- generic f16 WMMA GEMM: C = A[MxK] * W[KxN] + bias ----------------
// A row-major f16, Wt = W transposed [N][K] f16. 64x64 output tile per 256-thread block.
// Full K=128 chunk staged per barrier via TDM (wave 0 issues tensor_load_to_lds for both
// tiles, waits TENSORcnt, then the block syncs) -> 8 WMMAs per wave per sync.
__global__ __launch_bounds__(256) void k_gemm(
    const h16* __restrict__ A, const h16* __restrict__ Wt, const float* __restrict__ bias,
    float* __restrict__ Cf, h16* __restrict__ Ch, int M, int N, int K, int relu) {
  __shared__ __align__(32) h16 As[64 * 136];
  __shared__ __align__(32) h16 Ws[64 * 136];
  (void)M;
  int t = threadIdx.x, lane = t & 31, w = t >> 5;
  int m0 = blockIdx.x * 64, n0 = blockIdx.y * 64;
  int m16 = (w & 3) * 16, n16 = (w >> 2) * 32;
  int l15 = lane & 15, lhi = lane >> 4;
  v8f acc0 = {}, acc1 = {};
  unsigned ldsA = (unsigned)(size_t)&As[0];
  unsigned ldsW = (unsigned)(size_t)&Ws[0];
  for (int k0 = 0; k0 < K; k0 += 128) {
    if (w == 0) {                     // uniform per-wave branch; TDM issued once
      tdm_load_tile(ldsA, (const void*)(A  + (size_t)m0 * K + k0), K);
      tdm_load_tile(ldsW, (const void*)(Wt + (size_t)n0 * K + k0), K);
      __builtin_amdgcn_s_wait_tensorcnt(0);
    }
    __syncthreads();
#pragma unroll
    for (int ks = 0; ks < 4; ++ks) {
      const int ko = ks * 32;
      v16h af = mk16(&As[(m16 + l15) * 136 + ko + lhi * 8],
                     &As[(m16 + l15) * 136 + ko + lhi * 8 + 16]);
      v16h b0 = mk16(&Ws[(n16 + l15) * 136 + ko + lhi * 8],
                     &Ws[(n16 + l15) * 136 + ko + lhi * 8 + 16]);
      v16h b1 = mk16(&Ws[(n16 + 16 + l15) * 136 + ko + lhi * 8],
                     &Ws[(n16 + 16 + l15) * 136 + ko + lhi * 8 + 16]);
      acc0 = wmma_f16(af, b0, acc0);
      acc1 = wmma_f16(af, b1, acc1);
    }
    __syncthreads();
  }
  int col0 = n0 + n16 + l15;
  float bi0 = bias[col0], bi1 = bias[col0 + 16];
#pragma unroll
  for (int r = 0; r < 8; ++r) {
    int row = m0 + m16 + lhi * 8 + r;
    float v0 = acc0[r] + bi0, v1 = acc1[r] + bi1;
    if (relu) { v0 = fmaxf(v0, 0.f); v1 = fmaxf(v1, 0.f); }
    size_t o = (size_t)row * N + col0;
    if (Cf) { Cf[o] = v0; Cf[o + 16] = v1; }
    if (Ch) { Ch[o] = (h16)v0; Ch[o + 16] = (h16)v1; }
  }
}

// ---------------- fused attention per (b,h): scores+bias+mask+softmax+P@V ----------------
__global__ __launch_bounds__(128) void k_attn(
    const h16* __restrict__ q, const h16* __restrict__ k, const h16* __restrict__ v,
    h16* __restrict__ ctx, const int* __restrict__ ccv,
    const int* __restrict__ spe, const float* __restrict__ spe_tab,
    const float* __restrict__ dist_m, const float* __restrict__ dbw, const float* __restrict__ dbb,
    float* __restrict__ attn_acc, int mode, int store_attn) {
  __shared__ __align__(32) h16 qs[64 * 40];
  __shared__ __align__(32) h16 ks[64 * 40];
  __shared__ __align__(32) h16 vt[32 * 72];   // v transposed: [dh][key]
  __shared__ __align__(32) h16 ps[64 * 72];   // probs, A-operand layout
  int bh = blockIdx.x, b = bh >> 2, h = bh & 3;
  int t = threadIdx.x, lane = t & 31, w = t >> 5;
  int lr = t >> 1, lc = (t & 1) * 16;
  if (lr < Sz) {
    size_t base = ((size_t)(b * Sz + lr)) * Dz + h * DHz + lc;
    const v8h* qp = (const v8h*)(q + base);
    *(v8h*)&qs[lr * 40 + lc] = qp[0]; *(v8h*)&qs[lr * 40 + lc + 8] = qp[1];
    const v8h* kp = (const v8h*)(k + base);
    *(v8h*)&ks[lr * 40 + lc] = kp[0]; *(v8h*)&ks[lr * 40 + lc + 8] = kp[1];
    const h16* vp = v + base;
#pragma unroll
    for (int j = 0; j < 16; ++j) vt[(lc + j) * 72 + lr] = vp[j];
  } else {
    v8h z = {};
    *(v8h*)&qs[lr * 40 + lc] = z; *(v8h*)&qs[lr * 40 + lc + 8] = z;
    *(v8h*)&ks[lr * 40 + lc] = z; *(v8h*)&ks[lr * 40 + lc + 8] = z;
#pragma unroll
    for (int j = 0; j < 16; ++j) vt[(lc + j) * 72 + lr] = (h16)0.f;
  }
  __syncthreads();

  int m16 = w * 16;
  int l15 = lane & 15, lhi = lane >> 4;
  // ---- scores: q(16x32) @ k^T(32x64), K=32 -> one WMMA per 16x16 tile ----
  v16h af = mk16(&qs[(m16 + l15) * 40 + lhi * 8], &qs[(m16 + l15) * 40 + lhi * 8 + 16]);
  v8f acc[4];
#pragma unroll
  for (int ti = 0; ti < 4; ++ti) {
    v16h bf = mk16(&ks[(ti * 16 + l15) * 40 + lhi * 8], &ks[(ti * 16 + l15) * 40 + lhi * 8 + 16]);
    v8f z = {};
    acc[ti] = wmma_f16(af, bf, z);
  }
  int cb = ccv[b];
  float dwv = 0.f, dbv = 0.f;
  if (mode == 1) { dwv = dbw[h]; dbv = dbb[h]; }
  const float rs = 0.17677669529663687f;  // 1/sqrt(32)
#pragma unroll
  for (int ti = 0; ti < 4; ++ti) {
#pragma unroll
    for (int r = 0; r < 8; ++r) {
      int row = m16 + lhi * 8 + r;
      int col = ti * 16 + l15;
      float s = acc[ti][r] * rs;
      if (row < Sz) {
        if (col < Sz) {
          size_t bi = (size_t)b * Sz * Sz + (size_t)row * Sz + col;
          s += (mode == 0) ? spe_tab[spe[bi] * Hz + h] : (dist_m[bi] * dwv + dbv);
          if (col >= cb) s += NEGV;       // padding mask on keys
        } else s = -1e30f;                // 50..63 tile pad
      } else s = 0.f;                     // garbage query rows, never stored
      acc[ti][r] = s;
    }
  }
  // ---- softmax per row: values live in lane groups {0..15} / {16..31} ----
#pragma unroll
  for (int r = 0; r < 8; ++r) {
    float m = acc[0][r];
#pragma unroll
    for (int ti = 1; ti < 4; ++ti) m = fmaxf(m, acc[ti][r]);
    m = fmaxf(m, __shfl_xor(m, 1, 32));
    m = fmaxf(m, __shfl_xor(m, 2, 32));
    m = fmaxf(m, __shfl_xor(m, 4, 32));
    m = fmaxf(m, __shfl_xor(m, 8, 32));
    float sum = 0.f;
#pragma unroll
    for (int ti = 0; ti < 4; ++ti) { float e = __expf(acc[ti][r] - m); acc[ti][r] = e; sum += e; }
    sum += __shfl_xor(sum, 1, 32);
    sum += __shfl_xor(sum, 2, 32);
    sum += __shfl_xor(sum, 4, 32);
    sum += __shfl_xor(sum, 8, 32);
    float inv = 1.f / sum;
#pragma unroll
    for (int ti = 0; ti < 4; ++ti) acc[ti][r] *= inv;
  }
  // stage probs to LDS in A-fragment-friendly row-major layout
#pragma unroll
  for (int ti = 0; ti < 4; ++ti)
#pragma unroll
    for (int r = 0; r < 8; ++r)
      ps[(m16 + lhi * 8 + r) * 72 + ti * 16 + l15] = (h16)acc[ti][r];
  // last-layer attention column sums for a2m/a3m
  if (store_attn) {
#pragma unroll
    for (int ti = 0; ti < 4; ++ti) {
      float part = 0.f;
#pragma unroll
      for (int r = 0; r < 8; ++r) { int row = m16 + lhi * 8 + r; if (row < Sz) part += acc[ti][r]; }
      part += __shfl_xor(part, 16, 32);
      int col = ti * 16 + l15;
      if (lane < 16 && col < Sz) atomicAdd(&attn_acc[b * Sz + col], part);
    }
  }
  __syncthreads();
  // ---- ctx = P(64x64) @ V(64x32): K=64 -> two chained WMMAs per tile ----
  v8f c0 = {}, c1 = {};
#pragma unroll
  for (int kc = 0; kc < 2; ++kc) {
    v16h pa = mk16(&ps[(m16 + l15) * 72 + kc * 32 + lhi * 8],
                   &ps[(m16 + l15) * 72 + kc * 32 + lhi * 8 + 16]);
    v16h b0 = mk16(&vt[l15 * 72 + kc * 32 + lhi * 8],
                   &vt[l15 * 72 + kc * 32 + lhi * 8 + 16]);
    v16h b1 = mk16(&vt[(16 + l15) * 72 + kc * 32 + lhi * 8],
                   &vt[(16 + l15) * 72 + kc * 32 + lhi * 8 + 16]);
    c0 = wmma_f16(pa, b0, c0);
    c1 = wmma_f16(pa, b1, c1);
  }
#pragma unroll
  for (int r = 0; r < 8; ++r) {
    int row = m16 + lhi * 8 + r;
    if (row < Sz) {
      size_t ob = ((size_t)(b * Sz + row)) * Dz + h * DHz;
      ctx[ob + l15] = (h16)c0[r];
      ctx[ob + 16 + l15] = (h16)c1[r];
    }
  }
}

// ---------------- residual + LayerNorm (writes f32 + f16 mirrors) ----------------
__global__ __launch_bounds__(128) void k_ln(
    const float* __restrict__ x, const float* __restrict__ y,
    const float* __restrict__ g, const float* __restrict__ be,
    float* __restrict__ ho, h16* __restrict__ hh) {
  __shared__ float rs1[4], rs2[4];
  size_t row = blockIdx.x; int t = threadIdx.x, lane = t & 31, w = t >> 5;
  float v = x[row * Dz + t] + y[row * Dz + t];
  float s = v, s2 = v * v;
  for (int ofs = 16; ofs > 0; ofs >>= 1) { s += __shfl_xor(s, ofs, 32); s2 += __shfl_xor(s2, ofs, 32); }
  if (lane == 0) { rs1[w] = s; rs2[w] = s2; }
  __syncthreads();
  float S1 = rs1[0] + rs1[1] + rs1[2] + rs1[3];
  float S2 = rs2[0] + rs2[1] + rs2[2] + rs2[3];
  float mean = S1 * (1.f / Dz);
  float var = S2 * (1.f / Dz) - mean * mean;
  float out = (v - mean) * rsqrtf(var + 1e-12f) * g[t] + be[t];
  ho[row * Dz + t] = out;
  hh[row * Dz + t] = (h16)out;
}

// ---------------- masked mean pool -> f16 [B,128] ----------------
__global__ __launch_bounds__(128) void k_pool(
    const float* __restrict__ hsrc, const int* __restrict__ ccv, h16* __restrict__ p16) {
  int b = blockIdx.x, t = threadIdx.x;
  int c = ccv[b];
  float a = 0.f;
  for (int s = 0; s < c; ++s) a += hsrc[((size_t)(b * Sz + s)) * Dz + t];
  a /= (float)(c > 0 ? c : 1);
  p16[b * Dz + t] = (h16)a;
}

__global__ void k_attn_final(const float* __restrict__ acc, float* __restrict__ dst) {
  int i = blockIdx.x * blockDim.x + threadIdx.x;
  if (i < Bz * Sz) dst[i] = acc[i] * (1.0f / (Sz * Hz));   // mean over q (S) then heads (H)
}

// =====================================================================
extern "C" void kernel_launch(void* const* d_in, const int* in_sizes, int n_in,
                              void* d_out, int out_size, void* d_ws, size_t ws_size,
                              hipStream_t stream) {
  (void)n_in; (void)out_size; (void)ws_size;
  // input indices (setup_inputs dict order, params flattened in insertion order)
  const float* emb2   = (const float*)d_in[0];
  const float* emb3   = (const float*)d_in[1];
  const float* dist3  = (const float*)d_in[2];
  const float* dist_m = (const float*)d_in[3];
  const float* deg_in = (const float*)d_in[4];
  const float* deg_out= (const float*)d_in[5];
  const float* spe_tab= (const float*)d_in[6];
  const float* dinw   = (const float*)d_in[7];
  const float* dbw    = (const float*)d_in[8];
  const float* dbb    = (const float*)d_in[9];
  const int*   batch  = (const int*)d_in[146];
  const int*   ideg   = (const int*)d_in[147];
  const int*   odeg   = (const int*)d_in[148];
  const int*   spe    = (const int*)d_in[149];
  int Nn = in_sizes[146];
  float* out = (float*)d_out;

  // workspace carve-up (256B aligned)
  char* wsp = (char*)d_ws; size_t off = 0;
  auto alloc = [&](size_t bytes) -> void* {
    off = (off + 255) & ~(size_t)255;
    void* p = wsp + off; off += bytes; return p;
  };
  int*   counts  = (int*)  alloc((size_t)Bz * 4);
  int*   starts  = (int*)  alloc((size_t)Bz * 4);
  int*   cc      = (int*)  alloc((size_t)Bz * 4);
  int*   nodemap = (int*)  alloc((size_t)MTOK * 4);
  float* attn_acc= (float*)alloc((size_t)Bz * Sz * 4);
  float* hf32    = (float*)alloc((size_t)MTOK * Dz * 4);
  float* yf32    = (float*)alloc((size_t)MTOK * Dz * 4);
  h16*   hf16    = (h16*)  alloc((size_t)MTOK * Dz * 2);
  h16*   qf16    = (h16*)  alloc((size_t)MTOK * Dz * 2);
  h16*   kf16    = (h16*)  alloc((size_t)MTOK * Dz * 2);
  h16*   vf16    = (h16*)  alloc((size_t)MTOK * Dz * 2);
  h16*   ff16    = (h16*)  alloc((size_t)MTOK * Iz * 2);
  h16*   pool16  = (h16*)  alloc((size_t)Bz * Dz * 2);
  h16*   t16     = (h16*)  alloc((size_t)Bz * Dz * 2);
  h16*   wbuf    = (h16*)  alloc((size_t)(8 * 131072 + 49152) * 2);
  h16*   ctx16   = qf16;   // safe alias: each (b,h) block reads exactly the slice it rewrites

  // --- scatter/pad bookkeeping ---
  hipMemsetAsync(counts, 0, (size_t)Bz * 4, stream);
  hipMemsetAsync(nodemap, 0xFF, (size_t)MTOK * 4, stream);  // -1
  k_count  <<<(Nn + 255) / 256, 256, 0, stream>>>(batch, counts, Nn);
  k_scan   <<<1, 1, 0, stream>>>(counts, starts, cc);
  k_nodemap<<<(Nn + 255) / 256, 256, 0, stream>>>(batch, starts, nodemap, Nn);

  // --- one-time weight conversion to transposed f16 ---
  for (int e = 0; e < 2; ++e) {
    int eb = (e == 0) ? 10 : 74;
    for (int l = 0; l < Lz; ++l) {
      int bi = eb + l * 16;
      h16* wb = wbuf + (size_t)(e * 4 + l) * 131072;
      k_cvt<<<64, 256, 0, stream>>>((const float*)d_in[bi + 0], wb + 0,     128, 128); // wq
      k_cvt<<<64, 256, 0, stream>>>((const float*)d_in[bi + 2], wb + 16384, 128, 128); // wk
      k_cvt<<<64, 256, 0, stream>>>((const float*)d_in[bi + 4], wb + 32768, 128, 128); // wv
      k_cvt<<<64, 256, 0, stream>>>((const float*)d_in[bi + 6], wb + 49152, 128, 128); // wo
      k_cvt<<<128, 256, 0, stream>>>((const float*)d_in[bi + 8],  wb + 65536, 128, 256); // w1
      k_cvt<<<128, 256, 0, stream>>>((const float*)d_in[bi + 10], wb + 98304, 256, 128); // w2
    }
  }
  h16* pw = wbuf + (size_t)8 * 131072;
  k_cvt<<<64, 256, 0, stream>>>((const float*)d_in[138], pw + 0,     128, 128);
  k_cvt<<<32, 256, 0, stream>>>((const float*)d_in[140], pw + 16384, 128, 64);
  k_cvt<<<64, 256, 0, stream>>>((const float*)d_in[142], pw + 24576, 128, 128);
  k_cvt<<<32, 256, 0, stream>>>((const float*)d_in[144], pw + 40960, 128, 64);

  // --- two encoders, sequential, sharing big buffers ---
  for (int e = 0; e < 2; ++e) {
    int eb = (e == 0) ? 10 : 74;
    hipMemsetAsync(attn_acc, 0, (size_t)Bz * Sz * 4, stream);
    if (e == 0)
      k_embed2d<<<MTOK, 128, 0, stream>>>(nodemap, emb2, deg_in, deg_out, ideg, odeg, hf32, hf16);
    else
      k_embed3d<<<MTOK, 128, 0, stream>>>(nodemap, emb3, dist3, dinw, hf32, hf16);

    for (int l = 0; l < Lz; ++l) {
      int bi = eb + l * 16;
      h16* wb = wbuf + (size_t)(e * 4 + l) * 131072;
      dim3 g2(MTOK / 64, 2), g4(MTOK / 64, 4);
      k_gemm<<<g2, 256, 0, stream>>>(hf16, wb + 0,     (const float*)d_in[bi + 1], nullptr, qf16, MTOK, 128, 128, 0);
      k_gemm<<<g2, 256, 0, stream>>>(hf16, wb + 16384, (const float*)d_in[bi + 3], nullptr, kf16, MTOK, 128, 128, 0);
      k_gemm<<<g2, 256, 0, stream>>>(hf16, wb + 32768, (const float*)d_in[bi + 5], nullptr, vf16, MTOK, 128, 128, 0);
      k_attn<<<Bz * Hz, 128, 0, stream>>>(qf16, kf16, vf16, ctx16, cc, spe, spe_tab,
                                          dist_m, dbw, dbb, attn_acc, e, (l == Lz - 1) ? 1 : 0);
      k_gemm<<<g2, 256, 0, stream>>>(ctx16, wb + 49152, (const float*)d_in[bi + 7], yf32, nullptr, MTOK, 128, 128, 0);
      k_ln  <<<MTOK, 128, 0, stream>>>(hf32, yf32, (const float*)d_in[bi + 12], (const float*)d_in[bi + 13], hf32, hf16);
      k_gemm<<<g4, 256, 0, stream>>>(hf16, wb + 65536, (const float*)d_in[bi + 9],  nullptr, ff16, MTOK, 256, 128, 1);
      k_gemm<<<g2, 256, 0, stream>>>(ff16, wb + 98304, (const float*)d_in[bi + 11], yf32, nullptr, MTOK, 128, 256, 0);
      k_ln  <<<MTOK, 128, 0, stream>>>(hf32, yf32, (const float*)d_in[bi + 14], (const float*)d_in[bi + 15], hf32, hf16);
    }
    // pooled projection head
    k_pool<<<Bz, 128, 0, stream>>>(hf32, cc, pool16);
    h16* pwe = pw + (size_t)e * 24576;
    dim3 gp1(Bz / 64, 2), gp2(Bz / 64, 1);
    k_gemm<<<gp1, 256, 0, stream>>>(pool16, pwe + 0,     (const float*)d_in[e == 0 ? 139 : 143], nullptr, t16, Bz, 128, 128, 1);
    k_gemm<<<gp2, 256, 0, stream>>>(t16,    pwe + 16384, (const float*)d_in[e == 0 ? 141 : 145], out + (size_t)e * Bz * 64, nullptr, Bz, 64, 128, 0);
    // attention score output
    k_attn_final<<<(Bz * Sz + 255) / 256, 256, 0, stream>>>(attn_acc, out + 2 * (size_t)Bz * 64 + (size_t)e * Bz * Sz);
  }
}
